// CausalSelfAttention_38766374814266
// MI455X (gfx1250) — matrix operand order repaired and checked by reading, not compiled, not run on hardware
//
#include <hip/hip_runtime.h>
#include <hip/hip_bf16.h>

#define EMBED 1024
#define NHEADS 16
#define HDIM 64
#define BB 2
#define NN 2048
#define BN (BB * NN)        // 4096 tokens
#define BH (BB * NHEADS)    // 32 (batch*heads)

typedef _Float16 f16;
typedef _Float16 v16h __attribute__((ext_vector_type(16)));
typedef _Float16 v8h  __attribute__((ext_vector_type(8)));
typedef float    v8f  __attribute__((ext_vector_type(8)));
typedef unsigned int v4u __attribute__((ext_vector_type(4)));
typedef int      v8i  __attribute__((ext_vector_type(8)));
typedef int      v4i  __attribute__((ext_vector_type(4)));

// ---------------------------------------------------------------------------
// WMMA helpers (CDNA5: D = A(16x32 f16) x B(32x16 f16) + C(16x16 f32))
// ---------------------------------------------------------------------------
__device__ __forceinline__ v8f wmma(v16h a, v16h b, v8f c) {
  return __builtin_amdgcn_wmma_f32_16x16x32_f16(
      /*neg_a=*/false, a, /*neg_b=*/false, b,
      /*c_mod=*/(short)0, c, /*reuse_a=*/false, /*reuse_b=*/false);
}

// A-fragment: lane holds row (lane&15); K = (lane>>4)*16 .. +15 contiguous,
// koff = (lane>>4)*8.
__device__ __forceinline__ v16h load_fragA(const f16* base, int koff) {
  v8h lo = *(const v8h*)(base + 2 * koff);
  v8h hi = *(const v8h*)(base + 2 * koff + 8);
  v16h r;
#pragma unroll
  for (int i = 0; i < 8; ++i) { r[i] = lo[i]; r[8 + i] = hi[i]; }
  return r;
}

// B-fragment: lane holds column (lane&15); K = (lane>>4)*16 .. +15 contiguous.
__device__ __forceinline__ v16h load_fragB(const f16* base, int hi16) {
  v8h lo = *(const v8h*)(base + hi16);
  v8h hi = *(const v8h*)(base + hi16 + 8);
  v16h r;
#pragma unroll
  for (int i = 0; i < 8; ++i) { r[i] = lo[i]; r[8 + i] = hi[i]; }
  return r;
}

// ---------------------------------------------------------------------------
// Kernel 1: LayerNorm (f32 in) -> f16 normalized activations
// ---------------------------------------------------------------------------
__global__ __launch_bounds__(256) void ln_kernel(
    const float* __restrict__ x, const float* __restrict__ gamma,
    const float* __restrict__ beta, f16* __restrict__ xn) {
  int row = blockIdx.x;
  int tid = threadIdx.x;
  const float* xr = x + (size_t)row * EMBED;
  float v[4], s = 0.f, ss = 0.f;
#pragma unroll
  for (int i = 0; i < 4; ++i) {
    float t = xr[tid + 256 * i];
    v[i] = t; s += t; ss += t * t;
  }
#pragma unroll
  for (int m = 1; m < 32; m <<= 1) {
    s  += __shfl_xor(s,  m, 32);
    ss += __shfl_xor(ss, m, 32);
  }
  __shared__ float red_s[8], red_ss[8];
  int wid = tid >> 5, lane = tid & 31;
  if (lane == 0) { red_s[wid] = s; red_ss[wid] = ss; }
  __syncthreads();
  if (wid == 0) {
    float a = (lane < 8) ? red_s[lane] : 0.f;
    float b = (lane < 8) ? red_ss[lane] : 0.f;
#pragma unroll
    for (int m = 1; m < 8; m <<= 1) {
      a += __shfl_xor(a, m, 32);
      b += __shfl_xor(b, m, 32);
    }
    if (lane == 0) { red_s[0] = a; red_ss[0] = b; }
  }
  __syncthreads();
  float mean = red_s[0] * (1.f / EMBED);
  float var  = red_ss[0] * (1.f / EMBED) - mean * mean;
  float rs = rsqrtf(var + 1e-5f);
  f16* xo = xn + (size_t)row * EMBED;
#pragma unroll
  for (int i = 0; i < 4; ++i) {
    int c = tid + 256 * i;
    xo[c] = (f16)((v[i] - mean) * rs * gamma[c] + beta[c]);
  }
}

// ---------------------------------------------------------------------------
// Kernel 2: f32 -> f16 weight conversion
// ---------------------------------------------------------------------------
__global__ __launch_bounds__(256) void cvt_kernel(
    const float* __restrict__ src, f16* __restrict__ dst, int n) {
  int i = blockIdx.x * 256 + threadIdx.x;
  if (i < n) dst[i] = (f16)src[i];
}

// ---------------------------------------------------------------------------
// Shared GEMM core: one wave computes a 32(M) x 64(N) f32 tile of A @ W.T.
// Software-pipelined: next k-step fragments are loaded before this step's
// WMMAs, so the matrix pipe never waits on loadcnt==0 in steady state.
// ---------------------------------------------------------------------------
__device__ __forceinline__ void gemm32x64(
    const f16* __restrict__ A, const f16* __restrict__ W,
    int m0, int n0, int K, v8f acc[8]) {
  int lane = threadIdx.x & 31;
  int j = lane & 15;
  int koff = (lane >> 4) * 8;
  int hi16 = (lane >> 4) * 16;
  const f16* a0r = A + (size_t)(m0 + j) * K;
  const f16* a1r = A + (size_t)(m0 + 16 + j) * K;
  const f16* w0 = W + (size_t)(n0 + j) * K;
  const f16* w1 = w0 + (size_t)16 * K;
  const f16* w2 = w0 + (size_t)32 * K;
  const f16* w3 = w0 + (size_t)48 * K;

  v16h ca0 = load_fragA(a0r, koff);
  v16h ca1 = load_fragA(a1r, koff);
  v16h cb0 = load_fragB(w0, hi16);
  v16h cb1 = load_fragB(w1, hi16);
  v16h cb2 = load_fragB(w2, hi16);
  v16h cb3 = load_fragB(w3, hi16);

  for (int k = 32; k < K; k += 32) {
    v16h na0 = load_fragA(a0r + k, koff);
    v16h na1 = load_fragA(a1r + k, koff);
    v16h nb0 = load_fragB(w0 + k, hi16);
    v16h nb1 = load_fragB(w1 + k, hi16);
    v16h nb2 = load_fragB(w2 + k, hi16);
    v16h nb3 = load_fragB(w3 + k, hi16);
    acc[0] = wmma(ca0, cb0, acc[0]);
    acc[1] = wmma(ca0, cb1, acc[1]);
    acc[2] = wmma(ca0, cb2, acc[2]);
    acc[3] = wmma(ca0, cb3, acc[3]);
    acc[4] = wmma(ca1, cb0, acc[4]);
    acc[5] = wmma(ca1, cb1, acc[5]);
    acc[6] = wmma(ca1, cb2, acc[6]);
    acc[7] = wmma(ca1, cb3, acc[7]);
    ca0 = na0; ca1 = na1;
    cb0 = nb0; cb1 = nb1; cb2 = nb2; cb3 = nb3;
  }
  acc[0] = wmma(ca0, cb0, acc[0]);
  acc[1] = wmma(ca0, cb1, acc[1]);
  acc[2] = wmma(ca0, cb2, acc[2]);
  acc[3] = wmma(ca0, cb3, acc[3]);
  acc[4] = wmma(ca1, cb0, acc[4]);
  acc[5] = wmma(ca1, cb1, acc[5]);
  acc[6] = wmma(ca1, cb2, acc[6]);
  acc[7] = wmma(ca1, cb3, acc[7]);
}

// ---------------------------------------------------------------------------
// Kernel 3: fused QKV projection.  blockIdx.z: 0=Q, 1=K, 2=V
//   Q,K -> head layout [BH][N][D] f16 ; V -> transposed [BH][D][N] f16
// ---------------------------------------------------------------------------
__global__ __launch_bounds__(256) void gemm_qkv_kernel(
    const f16* __restrict__ xn,
    const f16* __restrict__ wq, const f16* __restrict__ wk, const f16* __restrict__ wv,
    const float* __restrict__ bq, const float* __restrict__ bk, const float* __restrict__ bv,
    f16* __restrict__ qh, f16* __restrict__ kh, f16* __restrict__ vt) {
  int w = threadIdx.x >> 5, lane = threadIdx.x & 31;
  int hi = lane >> 4, j = lane & 15;
  int m0 = blockIdx.x * 256 + w * 32;
  int n0 = blockIdx.y * 64;
  int z = blockIdx.z;
  const f16*   W    = (z == 0) ? wq : (z == 1) ? wk : wv;
  const float* bias = (z == 0) ? bq : (z == 1) ? bk : bv;

  v8f acc[8];
#pragma unroll
  for (int a = 0; a < 8; ++a) { v8f zr = {}; acc[a] = zr; }
  gemm32x64(xn, W, m0, n0, EMBED, acc);

#pragma unroll
  for (int mb = 0; mb < 2; ++mb) {
#pragma unroll
    for (int a = 0; a < 4; ++a) {
      int o = n0 + a * 16 + j;
      float bv_ = bias[o];
      int h = o >> 6, d = o & 63;
#pragma unroll
      for (int r = 0; r < 8; ++r) {
        int t = m0 + mb * 16 + r + 8 * hi;
        int b = t >> 11, n = t & (NN - 1);
        int bh = b * NHEADS + h;
        f16 hv = (f16)(acc[mb * 4 + a][r] + bv_);
        if (z == 2)
          vt[((size_t)bh * HDIM + d) * NN + n] = hv;
        else if (z == 0)
          qh[((size_t)bh * NN + n) * HDIM + d] = hv;
        else
          kh[((size_t)bh * NN + n) * HDIM + d] = hv;
      }
    }
  }
}

// ---------------------------------------------------------------------------
// Kernel 4: flash attention per (b,h).  Wave = 16 query rows, block = 128.
// Rel-bias table DMA'd into LDS via the Tensor Data Mover; K/V fragments
// are loaded up-front each key-tile so V completes under the softmax VALU.
// ---------------------------------------------------------------------------
__global__ __launch_bounds__(256) void attn_kernel(
    const f16* __restrict__ qh, const f16* __restrict__ kh,
    const f16* __restrict__ vt, const float* __restrict__ rel_bias,
    f16* __restrict__ oh) {
  __shared__ float s_rel[32 * 32];          // this head's 32x32 bias table
  __shared__ f16   s_p[8][16][32];          // per-wave P staging

  int bh = blockIdx.y;
  int b = bh / NHEADS, h = bh % NHEADS;
  int tid = threadIdx.x;
  const float* rb = rel_bias + (size_t)h * 32 * 32;

#if __has_builtin(__builtin_amdgcn_tensor_load_to_lds) && __has_builtin(__builtin_amdgcn_s_wait_tensorcnt)
  // TDM: one wave issues a 1024-element (1 row) f32 tile descriptor.
  // This toolchain exposes the 6-arg builtin (g0, g1, g2, g3, g4, cpol).
  if (tid < 32) {
    unsigned long long ga = (unsigned long long)(uintptr_t)rb;
    unsigned int lds = (unsigned int)(uintptr_t)(void*)s_rel;  // low 32b = LDS addr
    v4u g0;
    g0[0] = 1u;                                   // count=1, user descriptor
    g0[1] = lds;                                  // lds_addr
    g0[2] = (unsigned int)(ga & 0xFFFFFFFFu);     // global_addr[31:0]
    g0[3] = (unsigned int)((ga >> 32) & 0x01FFFFFFu) | (2u << 30);  // addr[56:32] | type=2
    v8i g1;
    g1[0] = (int)(2u << 16);                      // workgroup_mask=0, data_size=4B
    g1[1] = (int)(1024u << 16);                   // tensor_dim0[15:0]=1024 (bits 63:48)
    g1[2] = (int)(1u << 16);                      // tensor_dim1[15:0]=1 (bits 95:80)
    g1[3] = (int)(1024u << 16);                   // tile_dim0=1024 (bits 127:112)
    g1[4] = 1;                                    // tile_dim1=1
    g1[5] = 1024;                                 // tensor_dim0_stride[31:0]
    g1[6] = 0;
    g1[7] = 0;
    v4i gz4 = {};                                 // groups 2/3 unused (<=2D)
    v8i gz8 = {};
    __builtin_amdgcn_tensor_load_to_lds(g0, g1, gz4, gz4, gz8, 0);
    __builtin_amdgcn_s_wait_tensorcnt((short)0);
  }
#else
  for (int i = tid; i < 1024; i += 256) s_rel[i] = rb[i];
#endif
  __syncthreads();

  int w = tid >> 5, lane = tid & 31;
  int hi = lane >> 4, j = lane & 15;
  int koff = hi * 8, hi16 = hi * 16;
  int q0 = blockIdx.x * 128 + w * 16;
  const float pscale = 31.0f / 2047.0f;     // align_corners bilinear position
  const float sscale = 0.125f;              // 1/sqrt(64)
  const float NEG_INF = -__builtin_inff();

  // loop-invariant Q A-fragments (K = d, 0..63)
  const f16* qrow = qh + ((size_t)bh * NN + (q0 + j)) * HDIM;
  v16h aq0 = load_fragA(qrow + 0,  koff);
  v16h aq1 = load_fragA(qrow + 32, koff);

  int i0n[8]; float fn[8], mrow[8], lp[8];
#pragma unroll
  for (int r = 0; r < 8; ++r) {
    int qn = q0 + r + 8 * hi;
    float pos = qn * pscale;
    int i0 = (int)pos; if (i0 > 30) i0 = 30;
    i0n[r] = i0; fn[r] = pos - (float)i0;
    mrow[r] = NEG_INF; lp[r] = 0.f;
  }
  v8f oacc[4];
#pragma unroll
  for (int a = 0; a < 4; ++a) { v8f zr = {}; oacc[a] = zr; }

  const f16* kbase = kh + (size_t)bh * NN * HDIM;
  const f16* vbase = vt + (size_t)bh * HDIM * NN;
  int ktmax = (q0 + 15) >> 5;               // causal: keys <= max query row

  for (int kt = 0; kt <= ktmax; ++kt) {
    int kcol0 = kt * 32;
    // ---- issue ALL K and V fragment loads first (V covered by softmax) ----
    const f16* krow0 = kbase + (size_t)(kcol0 + j) * HDIM;
    const f16* krow1 = kbase + (size_t)(kcol0 + 16 + j) * HDIM;
    v16h bk00 = load_fragB(krow0 + 0,  hi16);
    v16h bk01 = load_fragB(krow0 + 32, hi16);
    v16h bk10 = load_fragB(krow1 + 0,  hi16);
    v16h bk11 = load_fragB(krow1 + 32, hi16);
    v16h bv0 = load_fragB(vbase + (size_t)(0  + j) * NN + kcol0, hi16);
    v16h bv1 = load_fragB(vbase + (size_t)(16 + j) * NN + kcol0, hi16);
    v16h bv2 = load_fragB(vbase + (size_t)(32 + j) * NN + kcol0, hi16);
    v16h bv3 = load_fragB(vbase + (size_t)(48 + j) * NN + kcol0, hi16);

    // ---- S = Q Kᵀ (16 x 32), two column halves ----
    v8f s01[2];
    {
      v8f s = {};
      s = wmma(aq0, bk00, s);
      s = wmma(aq1, bk01, s);
      s01[0] = s;
      v8f t = {};
      t = wmma(aq0, bk10, t);
      t = wmma(aq1, bk11, t);
      s01[1] = t;
    }
    // ---- scale + rel-bias interp + causal mask + online softmax ----
    float p[2][8], rmax[8];
#pragma unroll
    for (int r = 0; r < 8; ++r) {
      int qn = q0 + r + 8 * hi;
      float rm = NEG_INF;
#pragma unroll
      for (int c = 0; c < 2; ++c) {
        int km = kcol0 + c * 16 + j;
        float sv = s01[c][r] * sscale;
        float relv = 0.f;
        if (qn != NN - 1 && km != NN - 1) {
          float posm = km * pscale;
          int i0m = (int)posm; if (i0m > 30) i0m = 30;
          float fm = posm - (float)i0m;
          int i0 = i0n[r]; float f = fn[r];
          float r00 = s_rel[i0 * 32 + i0m],       r01 = s_rel[i0 * 32 + i0m + 1];
          float r10 = s_rel[(i0 + 1) * 32 + i0m], r11 = s_rel[(i0 + 1) * 32 + i0m + 1];
          relv = (1.f - f) * ((1.f - fm) * r00 + fm * r01) +
                 f        * ((1.f - fm) * r10 + fm * r11);
        }
        sv += relv;
        if (km > qn) sv = NEG_INF;
        p[c][r] = sv;
        rm = fmaxf(rm, sv);
      }
      rmax[r] = rm;
    }
#pragma unroll
    for (int r = 0; r < 8; ++r) {
      float rm = rmax[r];
#pragma unroll
      for (int m = 1; m < 16; m <<= 1) rm = fmaxf(rm, __shfl_xor(rm, m, 32));
      float mnew = fmaxf(mrow[r], rm);
      float alpha = __expf(mrow[r] - mnew);
      mrow[r] = mnew;
      float ps = 0.f;
#pragma unroll
      for (int c = 0; c < 2; ++c) {
        float e = __expf(p[c][r] - mnew);
        p[c][r] = e; ps += e;
      }
      lp[r] = lp[r] * alpha + ps;    // per-lane partial row sum (reduced at end)
#pragma unroll
      for (int a = 0; a < 4; ++a) oacc[a][r] *= alpha;
    }
    // ---- transpose P via per-wave LDS (C-layout -> A-layout) ----
#pragma unroll
    for (int r = 0; r < 8; ++r) {
#pragma unroll
      for (int c = 0; c < 2; ++c)
        s_p[w][r + 8 * hi][c * 16 + j] = (f16)p[c][r];
    }
    asm volatile("s_wait_dscnt 0" ::: "memory");   // within-wave LDS RAW
    v16h pa = load_fragA(&s_p[w][j][0], koff);
    // ---- O += P V  (V fragments already resident) ----
    oacc[0] = wmma(pa, bv0, oacc[0]);
    oacc[1] = wmma(pa, bv1, oacc[1]);
    oacc[2] = wmma(pa, bv2, oacc[2]);
    oacc[3] = wmma(pa, bv3, oacc[3]);
  }

  // ---- normalize and store merged-head output [BN][EMBED] f16 ----
  float inv[8];
#pragma unroll
  for (int r = 0; r < 8; ++r) {
    float ls = lp[r];
#pragma unroll
    for (int m = 1; m < 16; m <<= 1) ls += __shfl_xor(ls, m, 32);
    inv[r] = 1.f / ls;
  }
#pragma unroll
  for (int a = 0; a < 4; ++a) {
    int d = a * 16 + j;
#pragma unroll
    for (int r = 0; r < 8; ++r) {
      int qn = q0 + r + 8 * hi;
      size_t t = (size_t)b * NN + qn;
      oh[t * EMBED + h * HDIM + d] = (f16)(oacc[a][r] * inv[r]);
    }
  }
}

// ---------------------------------------------------------------------------
// Kernel 5: output projection  out = O @ Wo.T + bo  (f32 output)
// ---------------------------------------------------------------------------
__global__ __launch_bounds__(256) void gemm_out_kernel(
    const f16* __restrict__ oh, const f16* __restrict__ wo,
    const float* __restrict__ bo, float* __restrict__ out) {
  int w = threadIdx.x >> 5, lane = threadIdx.x & 31;
  int hi = lane >> 4, j = lane & 15;
  int m0 = blockIdx.x * 256 + w * 32;
  int n0 = blockIdx.y * 64;
  v8f acc[8];
#pragma unroll
  for (int a = 0; a < 8; ++a) { v8f zr = {}; acc[a] = zr; }
  gemm32x64(oh, wo, m0, n0, EMBED, acc);
#pragma unroll
  for (int mb = 0; mb < 2; ++mb) {
#pragma unroll
    for (int a = 0; a < 4; ++a) {
      int o = n0 + a * 16 + j;
      float bv_ = bo[o];
#pragma unroll
      for (int r = 0; r < 8; ++r) {
        int t = m0 + mb * 16 + r + 8 * hi;
        out[(size_t)t * EMBED + o] = acc[mb * 4 + a][r] + bv_;
      }
    }
  }
}

// ---------------------------------------------------------------------------
extern "C" void kernel_launch(void* const* d_in, const int* in_sizes, int n_in,
                              void* d_out, int out_size, void* d_ws, size_t ws_size,
                              hipStream_t stream) {
  const float* x     = (const float*)d_in[0];
  const float* Wq    = (const float*)d_in[1];
  const float* bq    = (const float*)d_in[2];
  const float* Wk    = (const float*)d_in[3];
  const float* bk    = (const float*)d_in[4];
  const float* Wv    = (const float*)d_in[5];
  const float* bv    = (const float*)d_in[6];
  const float* Wo    = (const float*)d_in[7];
  const float* bo    = (const float*)d_in[8];
  const float* gamma = (const float*)d_in[9];
  const float* beta  = (const float*)d_in[10];
  const float* rel   = (const float*)d_in[11];
  float* out = (float*)d_out;

  char* ws = (char*)d_ws;
  size_t off = 0;
  auto alloc = [&](size_t bytes) -> void* {
    void* p = ws + off;
    off += (bytes + 255) & ~(size_t)255;
    return p;
  };
  f16* xn   = (f16*)alloc((size_t)BN * EMBED * sizeof(f16));     // 8 MB
  f16* wq_h = (f16*)alloc((size_t)EMBED * EMBED * sizeof(f16));  // 2 MB
  f16* wk_h = (f16*)alloc((size_t)EMBED * EMBED * sizeof(f16));
  f16* wv_h = (f16*)alloc((size_t)EMBED * EMBED * sizeof(f16));
  f16* wo_h = (f16*)alloc((size_t)EMBED * EMBED * sizeof(f16));
  f16* qh   = (f16*)alloc((size_t)BH * NN * HDIM * sizeof(f16)); // 8 MB
  f16* kh   = (f16*)alloc((size_t)BH * NN * HDIM * sizeof(f16));
  f16* vt   = (f16*)alloc((size_t)BH * HDIM * NN * sizeof(f16));
  f16* oh   = (f16*)alloc((size_t)BN * EMBED * sizeof(f16));     // 8 MB

  // 1) LayerNorm -> f16
  ln_kernel<<<BN, 256, 0, stream>>>(x, gamma, beta, xn);

  // 2) weights -> f16
  int wn = EMBED * EMBED;
  cvt_kernel<<<(wn + 255) / 256, 256, 0, stream>>>(Wq, wq_h, wn);
  cvt_kernel<<<(wn + 255) / 256, 256, 0, stream>>>(Wk, wk_h, wn);
  cvt_kernel<<<(wn + 255) / 256, 256, 0, stream>>>(Wv, wv_h, wn);
  cvt_kernel<<<(wn + 255) / 256, 256, 0, stream>>>(Wo, wo_h, wn);

  // 3) fused QKV projection (z selects Q/K/V)
  dim3 gq(BN / 256, EMBED / 64, 3);
  gemm_qkv_kernel<<<gq, 256, 0, stream>>>(xn, wq_h, wk_h, wv_h,
                                          bq, bk, bv, qh, kh, vt);

  // 4) flash attention with rel-bias + causal mask
  dim3 ga(NN / 128, BH, 1);
  attn_kernel<<<ga, 256, 0, stream>>>(qh, kh, vt, rel, oh);

  // 5) output projection -> f32
  dim3 go(BN / 256, EMBED / 64, 1);
  gemm_out_kernel<<<go, 256, 0, stream>>>(oh, wo_h, bo, out);
}
